// DeepsetMLP_1546188226863
// MI455X (gfx1250) — matrix operand order repaired and checked by reading, not compiled, not run on hardware
//
#include <hip/hip_runtime.h>
#include <cstdint>
#include <cstddef>

#define N_BS   8192
#define NA     3
#define NE     5
#define HD     256
#define RA_CNT (N_BS*NA)        /* 24576 agent rows            */
#define RE_CNT (RA_CNT*NE)      /* 122880 (b,a,i) rows         */
#define RP_CNT (RE_CNT*NE)      /* 614400 (b,a,i,j) pair rows  */
#define EPSF   1e-7f

typedef _Float16 v16h  __attribute__((ext_vector_type(16)));
typedef float    v8f   __attribute__((ext_vector_type(8)));
typedef unsigned u32x4 __attribute__((ext_vector_type(4)));

union Frag16 { u32x4 q[2]; v16h h; };   // 8 VGPRs worth of f16 fragment
union H2     { _Float16 h[2]; unsigned u; };

#define ASYNC_WAIT_8() asm volatile("s_wait_asynccnt 0x8" ::: "memory")
#define ASYNC_WAIT_0() asm volatile("s_wait_asynccnt 0x0" ::: "memory")

// one async 16B global->LDS copy; IOFFSET applies to both LDS and global addr
template<int OFF>
__device__ __forceinline__ void async_b128(unsigned la, uint64_t ga) {
  asm volatile("global_load_async_to_lds_b128 %0, %1, off offset:%2"
               :: "v"(la), "v"(ga), "i"(OFF) : "memory");
}

// ---------------------------------------------------------------------------
// Weight repack into WMMA B-fragment layout.
// B 32x16 f16 as packed here (wave32, contiguous k like A): VGPR v (0..7), lane l: n = l&15, half = l>>4,
// holds K = 2v + 16*half (lo 16b) and K+1 (hi 16b).
// Memory layout: [tileIdx][lane][v] dwords -> per-lane 8 contiguous dwords.
// Each 16-nt "slice" is 16 KB, staged into LDS by the GEMM kernel.
// ---------------------------------------------------------------------------
__global__ void pack_w1(const float* __restrict__ aW1, const float* __restrict__ eW1,
                        const float* __restrict__ lW1, unsigned* __restrict__ out) {
  int tid = blockIdx.x*blockDim.x + threadIdx.x;
  if (tid >= 3*16*32*8) return;
  int v = tid & 7, lane = (tid >> 3) & 31, nt = (tid >> 8) & 15, g = tid >> 12;
  int half = lane >> 4, n = nt*16 + (lane & 15);
  int K0 = 2*v + 16*half;
  const float* W = (g==0) ? aW1 : ((g==1) ? eW1 : lW1);
  H2 t;
  t.h[0] = (K0   < 9) ? (_Float16)W[(size_t)K0*HD + n]     : (_Float16)0.f;
  t.h[1] = (K0+1 < 9) ? (_Float16)W[(size_t)(K0+1)*HD + n] : (_Float16)0.f;
  out[tid] = t.u;
}

__global__ void pack_w2(const float* __restrict__ aW2, const float* __restrict__ eW2,
                        const float* __restrict__ lW2, unsigned* __restrict__ out) {
  int tid = blockIdx.x*blockDim.x + threadIdx.x;
  if (tid >= 3*8*16*32*8) return;
  int v = tid & 7, lane = (tid >> 3) & 31, nt = (tid >> 8) & 15;
  int kk = (tid >> 12) & 7, g = tid >> 15;
  int half = lane >> 4, n = nt*16 + (lane & 15);
  int K0 = kk*32 + 2*v + 16*half;                       // always < 256
  const float* W = (g==0) ? aW2 : ((g==1) ? eW2 : lW2);
  H2 t;
  t.h[0] = (_Float16)W[(size_t)K0*HD + n];
  t.h[1] = (_Float16)W[(size_t)(K0+1)*HD + n];
  out[tid] = t.u;
}

__global__ void pack_o1(const float* __restrict__ o1W, unsigned* __restrict__ out) {
  int tid = blockIdx.x*blockDim.x + threadIdx.x;
  if (tid >= 9*16*32*8) return;
  int v = tid & 7, lane = (tid >> 3) & 31, nt = (tid >> 8) & 15, kk = tid >> 12;
  int half = lane >> 4, n = nt*16 + (lane & 15);
  int K0 = kk*32 + 2*v + 16*half;
  H2 t;
  t.h[0] = (K0   < 259) ? (_Float16)o1W[(size_t)K0*HD + n]     : (_Float16)0.f;
  t.h[1] = (K0+1 < 259) ? (_Float16)o1W[(size_t)(K0+1)*HD + n] : (_Float16)0.f;
  out[tid] = t.u;
}

// combined layer-2 bias: 2*ally + enemy + 2*landmark (sum over j of b2_g(j))
__global__ void make_mbias(const float* __restrict__ ab2, const float* __restrict__ eb2,
                           const float* __restrict__ lb2, float* __restrict__ mb) {
  int n = threadIdx.x;
  mb[n] = 2.f*ab2[n] + eb2[n] + 2.f*lb2[n];
}

// ---------------------------------------------------------------------------
// Featurization: one thread per (b,a,i,j) pair -> 9 features, f16, padded to 32
// entity order: 0,1 = ally pos ob[8:12]; 2 = enemy ob[12:14]; 3,4 = landmark ob[4:8]
// vel: 0,1 = ob[14:18]; 2 = ob[18:20]; 3,4 = -ob[0:2]
// ---------------------------------------------------------------------------
__device__ __forceinline__ void get_ent(const float* o, int e,
                                        float& px, float& py, float& vx, float& vy) {
  if      (e == 0) { px = o[8];  py = o[9];  vx = o[14]; vy = o[15]; }
  else if (e == 1) { px = o[10]; py = o[11]; vx = o[16]; vy = o[17]; }
  else if (e == 2) { px = o[12]; py = o[13]; vx = o[18]; vy = o[19]; }
  else if (e == 3) { px = o[4];  py = o[5];  vx = -o[0]; vy = -o[1]; }
  else             { px = o[6];  py = o[7];  vx = -o[0]; vy = -o[1]; }
}

__global__ void featurize(const float* __restrict__ ob, const float* __restrict__ act,
                          _Float16* __restrict__ feat, float* __restrict__ actnorm,
                          float* __restrict__ actangle) {
  int tid = blockIdx.x*blockDim.x + threadIdx.x;
  if (tid >= RP_CNT) return;
  int j  = tid % NE;
  int R  = tid / NE;            // (b*3+a)*5 + i
  int i  = R % NE;
  int ra = R / NE;              // b*3+a
  const float* o = ob + (size_t)ra*20;
  float pxi,pyi,vxi,vyi, pxj,pyj,vxj,vyj;
  get_ent(o, i, pxi, pyi, vxi, vyi);
  get_ent(o, j, pxj, pyj, vxj, vyj);
  float di  = sqrtf(pxi*pxi + pyi*pyi);
  float sti = pyi/(di + EPSF), cti = pxi/(di + EPSF);
  float dj  = sqrtf(pxj*pxj + pyj*pyj);
  float stj = pyj/(dj + EPSF), ctj = pxj/(dj + EPSF);
  float vnj = sqrtf(vxj*vxj + vyj*vyj);
  float saj = vyj/(vnj + EPSF), caj = vxj/(vnj + EPSF);
  const float coeff = -0.02f;                      // -0.5/(5^2), offsets 0,5,10
  float f0 = expf(coeff * dj * dj);
  float f1 = expf(coeff * (dj-5.f)*(dj-5.f));
  float f2 = expf(coeff * (dj-10.f)*(dj-10.f));
  float sAB = stj*cti - ctj*sti;
  float cAB = ctj*cti + stj*sti;
  float sCB = saj*cti - caj*sti;
  float cCB = caj*cti + saj*sti;
  float rvx = vxj - vxi, rvy = vyj - vyi;
  _Float16* f = feat + ((size_t)j*RE_CNT + R)*32;
  f[0]=(_Float16)f0;  f[1]=(_Float16)f1;  f[2]=(_Float16)f2;
  f[3]=(_Float16)sAB; f[4]=(_Float16)cAB; f[5]=(_Float16)sCB; f[6]=(_Float16)cCB;
  f[7]=(_Float16)rvx; f[8]=(_Float16)rvy;
  #pragma unroll
  for (int k = 9; k < 32; ++k) f[k] = (_Float16)0.f;
  if (j == 0) {
    float ax = act[(size_t)ra*2], ay = act[(size_t)ra*2 + 1];
    float an = sqrtf(ax*ax + ay*ay);
    float sb = ay/(an + EPSF), cb = ax/(an + EPSF);
    actangle[(size_t)R*2]     = sb*cti - cb*sti;
    actangle[(size_t)R*2 + 1] = cb*cti + sb*sti;
    if (i == 0) actnorm[ra] = an;
  }
}

// ---------------------------------------------------------------------------
// Fused WMMA kernel: 1 wave handles 16 (b,a,i) rows, full pipeline.
// B slices (16 KB each) are staged block-cooperatively into LDS via
// GLOBAL_LOAD_ASYNC_TO_LDS_B128 (ASYNCcnt), double buffered so the async DMA
// overlaps the WMMAs. 4x reduction in L2 weight traffic vs per-wave loads.
// A 16x32 f16 layout: lane l: M = l&15, half = l>>4; VGPR v pairs at
// K = 2v + 16*half (k order used here, same as B) -> dwords {8h..8h+3, 8h+4..8h+7}.
// C/D f32: VGPR vv: M = vv + 8*half, N = lane&15.
// ---------------------------------------------------------------------------
__global__ __launch_bounds__(128) void fused_mlp(
    const _Float16* __restrict__ feat,
    const unsigned* __restrict__ w1p, const unsigned* __restrict__ w2p,
    const unsigned* __restrict__ o1p,
    const float* __restrict__ ab1, const float* __restrict__ eb1,
    const float* __restrict__ lb1,
    const float* __restrict__ mbias, const float* __restrict__ actnorm,
    const float* __restrict__ actangle,
    const float* __restrict__ o1b, const float* __restrict__ o2W,
    float* __restrict__ qrow, float* __restrict__ xout) {
  __shared__ _Float16 lds[4][16][304];     // per-wave activation tiles
  __shared__ u32x4    bstage[2][1024];     // 2 x 16KB B-slice staging
  const int tid  = threadIdx.x;            // 0..127
  const int w    = tid >> 5;
  const int lane = tid & 31;
  const int m    = lane & 15;
  const int half = lane >> 4;
  const int R0   = (blockIdx.x*4 + w) * 16;
  const v8f vz   = {0.f,0.f,0.f,0.f,0.f,0.f,0.f,0.f};

  // cooperative async stage of one 16KB slice: 128 threads x 8 x b128,
  // single base address per thread + immediate IOFFSET (applies to LDS+global)
  auto issue_slice = [&](const unsigned* src, int buf) {
    unsigned la = (unsigned)(uintptr_t)&bstage[buf][0] + (unsigned)(tid*16);
    uint64_t ga = (uint64_t)(uintptr_t)src + (uint64_t)(tid*16);
    async_b128<0>(la, ga);
    async_b128<2048>(la, ga);
    async_b128<4096>(la, ga);
    async_b128<6144>(la, ga);
    async_b128<8192>(la, ga);
    async_b128<10240>(la, ga);
    async_b128<12288>(la, ga);
    async_b128<14336>(la, ga);
  };

  v8f acc[16];
  #pragma unroll
  for (int nt = 0; nt < 16; ++nt) acc[nt] = vz;

  _Float16* Hrow = &lds[w][m][0];

  for (int j = 0; j < NE; ++j) {
    const int g = (j < 2) ? 0 : ((j == 2) ? 1 : 2);
    const float* b1 = (g == 0) ? ab1 : ((g == 1) ? eb1 : lb1);
    // stage layer-1 B slice (16KB) into buf0
    issue_slice(w1p + g*4096, 0);
    // layer-1 A fragment (16 rows x K32, K>=9 zero padded) - overlaps the DMA
    Frag16 a1;
    const u32x4* fp4 = (const u32x4*)(feat + ((size_t)j*RE_CNT + R0 + m)*32);
    a1.q[0] = fp4[2*half];
    a1.q[1] = fp4[2*half + 1];
    ASYNC_WAIT_0();
    __syncthreads();
    const unsigned* w2g = w2p + g*(8*16*32*8);
    issue_slice(w2g, 1);                       // prefetch layer-2 slice 0 -> buf1
    #pragma unroll
    for (int nt = 0; nt < 16; ++nt) {
      Frag16 bf;
      const u32x4* bp = &bstage[0][(nt*32 + lane)*2];
      bf.q[0] = bp[0]; bf.q[1] = bp[1];
      v8f c = __builtin_amdgcn_wmma_f32_16x16x32_f16(false, a1.h, false, bf.h,
                                                     (short)0, vz, false, false);
      float bias = b1[nt*16 + m];
      #pragma unroll
      for (int vv = 0; vv < 8; ++vv) {
        float y = c[vv] + bias;
        y = (y > 0.f) ? y : 0.01f*y;                 // leaky_relu
        lds[w][vv + 8*half][nt*16 + m] = (_Float16)y;
      }
    }
    __syncthreads();   // H1 visible; buf0 free for reuse
    // layer-2: H1 (16x256) @ W2_g, accumulate over j; double-buffered B slices
    for (int kk = 0; kk < 8; ++kk) {
      const int cur = (kk + 1) & 1;                  // kk=0 -> buf1
      if (kk < 7) issue_slice(w2g + (kk+1)*4096, kk & 1);
      if (kk < 7) { ASYNC_WAIT_8(); } else { ASYNC_WAIT_0(); }
      __syncthreads();
      Frag16 a2;
      a2.q[0] = *(const u32x4*)((const char*)Hrow + kk*64 + half*32);
      a2.q[1] = *(const u32x4*)((const char*)Hrow + kk*64 + 16 + half*32);
      #pragma unroll
      for (int nt = 0; nt < 16; ++nt) {
        Frag16 bf;
        const u32x4* bp = &bstage[cur][(nt*32 + lane)*2];
        bf.q[0] = bp[0]; bf.q[1] = bp[1];
        acc[nt] = __builtin_amdgcn_wmma_f32_16x16x32_f16(false, a2.h, false, bf.h,
                                                         (short)0, acc[nt], false, false);
      }
      __syncthreads();                               // buf[cur] free for reuse
    }
  }

  // prefetch output-layer slice 0 -> buf0 (both buffers quiesced above)
  issue_slice(o1p, 0);

  // final-layer A = relu([m + sum(b2), act_norm, act_angle, 0-pad]) -> LDS f16
  #pragma unroll
  for (int nt = 0; nt < 16; ++nt) {
    float mb = mbias[nt*16 + m];
    #pragma unroll
    for (int vv = 0; vv < 8; ++vv) {
      float y = acc[nt][vv] + mb;
      lds[w][vv + 8*half][nt*16 + m] = (_Float16)fmaxf(y, 0.f);
      acc[nt][vv] = 0.f;                             // reuse as output accumulator
    }
  }
  if (half == 0) {
    int Rg = R0 + m;
    lds[w][m][256] = (_Float16)fmaxf(actnorm[Rg/NE], 0.f);
    lds[w][m][257] = (_Float16)fmaxf(actangle[(size_t)Rg*2], 0.f);
    lds[w][m][258] = (_Float16)fmaxf(actangle[(size_t)Rg*2 + 1], 0.f);
    #pragma unroll
    for (int k = 259; k < 288; ++k) lds[w][m][k] = (_Float16)0.f;
  }
  __syncthreads();

  // output layer-1: (16x288) @ o1W(288x256 padded), double-buffered B slices
  for (int kk = 0; kk < 9; ++kk) {
    const int cur = kk & 1;                          // slice kk lives in buf[kk&1]
    if (kk < 8) issue_slice(o1p + (kk+1)*4096, (kk+1) & 1);
    if (kk < 8) { ASYNC_WAIT_8(); } else { ASYNC_WAIT_0(); }
    __syncthreads();
    Frag16 a3;
    a3.q[0] = *(const u32x4*)((const char*)Hrow + kk*64 + half*32);
    a3.q[1] = *(const u32x4*)((const char*)Hrow + kk*64 + 16 + half*32);
    #pragma unroll
    for (int nt = 0; nt < 16; ++nt) {
      Frag16 bf;
      const u32x4* bp = &bstage[cur][(nt*32 + lane)*2];
      bf.q[0] = bp[0]; bf.q[1] = bp[1];
      acc[nt] = __builtin_amdgcn_wmma_f32_16x16x32_f16(false, a3.h, false, bf.h,
                                                       (short)0, acc[nt], false, false);
    }
    __syncthreads();
  }

  // epilogue: bias + relu -> x (fp32 out), dot with o2W -> per-row q partial
  float p[8];
  #pragma unroll
  for (int vv = 0; vv < 8; ++vv) p[vv] = 0.f;
  #pragma unroll
  for (int nt = 0; nt < 16; ++nt) {
    float bias = o1b[nt*16 + m];
    float wq   = o2W[nt*16 + m];
    #pragma unroll
    for (int vv = 0; vv < 8; ++vv) {
      float y = fmaxf(acc[nt][vv] + bias, 0.f);
      xout[(size_t)(R0 + vv + 8*half)*HD + nt*16 + m] = y;
      p[vv] += y * wq;
    }
  }
  #pragma unroll
  for (int vv = 0; vv < 8; ++vv) {
    float s = p[vv];
    s += __shfl_xor(s, 1, 32);
    s += __shfl_xor(s, 2, 32);
    s += __shfl_xor(s, 4, 32);
    s += __shfl_xor(s, 8, 32);
    if (m == 0) qrow[R0 + vv + 8*half] = s;          // lanes 0 & 16 cover 16 rows
  }
}

__global__ void qfinal(const float* __restrict__ qrow, const float* __restrict__ o2b,
                       float* __restrict__ q) {
  int t = blockIdx.x*blockDim.x + threadIdx.x;
  if (t >= RA_CNT) return;
  float s = 0.f;
  #pragma unroll
  for (int i = 0; i < NE; ++i) s += qrow[(size_t)t*NE + i];
  q[t] = s * 0.2f + o2b[0];
}

// ---------------------------------------------------------------------------
extern "C" void kernel_launch(void* const* d_in, const int* in_sizes, int n_in,
                              void* d_out, int out_size, void* d_ws, size_t ws_size,
                              hipStream_t stream) {
  const float* ob  = (const float*)d_in[0];
  const float* act = (const float*)d_in[2];            // d_in[1] hidden_state unused
  const float* aW1 = (const float*)d_in[3];  const float* ab1 = (const float*)d_in[4];
  const float* aW2 = (const float*)d_in[5];  const float* ab2 = (const float*)d_in[6];
  const float* eW1 = (const float*)d_in[7];  const float* eb1 = (const float*)d_in[8];
  const float* eW2 = (const float*)d_in[9];  const float* eb2 = (const float*)d_in[10];
  const float* lW1 = (const float*)d_in[11]; const float* lb1 = (const float*)d_in[12];
  const float* lW2 = (const float*)d_in[13]; const float* lb2 = (const float*)d_in[14];
  const float* o1W = (const float*)d_in[15]; const float* o1b = (const float*)d_in[16];
  const float* o2W = (const float*)d_in[17]; const float* o2b = (const float*)d_in[18];

  char* ws = (char*)d_ws;
  size_t off = 0;
  _Float16* feat = (_Float16*)(ws + off); off += (size_t)NE*RE_CNT*32*2;  // 39.3 MB
  unsigned* w1p  = (unsigned*)(ws + off); off += (size_t)3*16*32*8*4;
  unsigned* w2p  = (unsigned*)(ws + off); off += (size_t)3*8*16*32*8*4;
  unsigned* o1p  = (unsigned*)(ws + off); off += (size_t)9*16*32*8*4;
  float*    mb   = (float*)(ws + off);    off += 256*4;
  float*    an   = (float*)(ws + off);    off += (size_t)RA_CNT*4;
  float*    aa   = (float*)(ws + off);    off += (size_t)RE_CNT*2*4;
  float*    qr   = (float*)(ws + off);    off += (size_t)RE_CNT*4;
  (void)in_sizes; (void)n_in; (void)out_size; (void)ws_size;

  float* q    = (float*)d_out;            // 24576
  float* xout = (float*)d_out + RA_CNT;   // 122880*256

  pack_w1   <<<48,  256, 0, stream>>>(aW1, eW1, lW1, w1p);
  pack_w2   <<<384, 256, 0, stream>>>(aW2, eW2, lW2, w2p);
  pack_o1   <<<144, 256, 0, stream>>>(o1W, o1p);
  make_mbias<<<1,   256, 0, stream>>>(ab2, eb2, lb2, mb);
  featurize <<<RP_CNT/256, 256, 0, stream>>>(ob, act, feat, an, aa);
  fused_mlp <<<RE_CNT/64, 128, 0, stream>>>(feat, w1p, w2p, o1p,
                                            ab1, eb1, lb1, mb, an, aa,
                                            o1b, o2W, qr, xout);
  qfinal    <<<RA_CNT/256, 256, 0, stream>>>(qr, o2b, q);
}
